// ULossRGBTopakgnc_26697516712402
// MI455X (gfx1250) — matrix operand order repaired and checked by reading, not compiled, not run on hardware
//
#include <hip/hip_runtime.h>
#include <hip/hip_bf16.h>

typedef __attribute__((ext_vector_type(16))) _Float16 v16h;
typedef __attribute__((ext_vector_type(8)))  float    v8f;

static constexpr int kB = 4, kH = 256, kW = 256, kHW = kH * kW;
static constexpr int kAng = 7, kN = 49, kCenter = 24, kCrop = 8, kInner = 240;
static constexpr float kEdgeC = 150.0f, kAlpha = 0.1f;

// workspace layout (float offsets)
static constexpr size_t CL_OFF = 0;
static constexpr size_t GA_OFF = (size_t)kB * kN * kHW;      // 12,845,056 floats
static constexpr size_t PD_OFF = 2 * GA_OFF;
static constexpr int    NBLK_P = 900;                        // 4*240*240/256
static constexpr size_t PE_OFF = PD_OFF + NBLK_P;

// ---------------------------------------------------------------------------
// Kernel 1: per-pixel, per-view photometric loss via bilinear warp gather.
// cl[b, n, h, w] = mean_c | bilinear(view n, (h,w) + pred*(du,dv)) - center |
// ---------------------------------------------------------------------------
__global__ void k_color_loss(const float* __restrict__ pred,
                             const float* __restrict__ x,
                             float* __restrict__ cl) {
  int idx = blockIdx.x * blockDim.x + threadIdx.x;   // 0 .. B*HW-1
  int b  = idx >> 16;
  int hw = idx & (kHW - 1);
  int h = hw >> 8, w = hw & 255;
  float p = pred[idx];

  const float* xc = x + ((size_t)(b * kN + kCenter) * kHW + hw) * 3;
  float c0 = xc[0], c1 = xc[1], c2 = xc[2];

  #pragma unroll 1
  for (int u = 0; u < kAng; ++u) {
    #pragma unroll 1
    for (int v = 0; v < kAng; ++v) {
      int n = u * kAng + v;
      // prefetch next view plane near this pixel (warp offset is <= 3*|p|)
      if (n + 1 < kN)
        __builtin_prefetch(x + ((size_t)(b * kN + n + 1) * kHW + hw) * 3, 0, 0);
      float du = (float)(u - 3), dv = (float)(v - 3);
      float cy = fminf(fmaxf((float)h + p * du, 0.f), (float)(kH - 1));
      float cx = fminf(fmaxf((float)w + p * dv, 0.f), (float)(kW - 1));
      float y0f = floorf(cy), x0f = floorf(cx);
      float ty = cy - y0f, tx = cx - x0f;
      int y0 = (int)y0f, x0 = (int)x0f;
      int y1 = min(y0 + 1, kH - 1), x1 = min(x0 + 1, kW - 1);
      size_t vb = (size_t)(b * kN + n) * kHW;
      const float* p00 = x + (vb + (size_t)y0 * kW + x0) * 3;
      const float* p01 = x + (vb + (size_t)y0 * kW + x1) * 3;
      const float* p10 = x + (vb + (size_t)y1 * kW + x0) * 3;
      const float* p11 = x + (vb + (size_t)y1 * kW + x1) * 3;
      float w00 = (1.f - ty) * (1.f - tx), w01 = (1.f - ty) * tx;
      float w10 = ty * (1.f - tx),         w11 = ty * tx;
      float r0 = w00 * p00[0] + w01 * p01[0] + w10 * p10[0] + w11 * p11[0];
      float r1 = w00 * p00[1] + w01 * p01[1] + w10 * p10[1] + w11 * p11[1];
      float r2 = w00 * p00[2] + w01 * p01[2] + w10 * p10[2] + w11 * p11[2];
      float acc = fabsf(r0 - c0) + fabsf(r1 - c1) + fabsf(r2 - c2);
      cl[(size_t)(b * kN + n) * kHW + hw] = acc * (1.f / 3.f);
    }
  }
}

// ---------------------------------------------------------------------------
// Kernel 2: angular Gaussian smoothing as a dense matmul on the tensor core:
//   gauss(64pad x Npix) = G(64x64, f16, in LDS) x CL(64pad x Npix, f16)
// One wave handles 16 pixel columns: 4 M-tiles x 2 K-steps of
// v_wmma_f32_16x16x32_f16. K order in A and B is contiguous, k = 16*group + j.
// All global addressing uses base pointer + constant row offsets (row stride
// kHW*4 = 256 KiB < 8 MiB IOFFSET range) so stores/loads fold into IOFFSET.
// ---------------------------------------------------------------------------
__global__ void k_gauss_wmma(const float* __restrict__ cl,
                             float* __restrict__ ga) {
  __shared__ _Float16 Gs[64 * 64];
  const float G3[3][3] = {{0.0751f, 0.1238f, 0.0751f},
                          {0.1238f, 0.2042f, 0.1238f},
                          {0.0751f, 0.1238f, 0.0751f}};
  int tid = threadIdx.x;
  for (int e = tid; e < 64 * 64; e += blockDim.x) {
    int n = e >> 6, m = e & 63;
    float val = 0.f;
    if (n < kN && m < kN) {
      int i = n / kAng, j = n % kAng;
      #pragma unroll
      for (int di = -1; di <= 1; ++di)
        #pragma unroll
        for (int dj = -1; dj <= 1; ++dj) {
          int ii = min(max(i + di, 0), kAng - 1);
          int jj = min(max(j + dj, 0), kAng - 1);
          if (ii * kAng + jj == m) val += G3[di + 1][dj + 1];  // edge padding
        }
    }
    Gs[e] = (_Float16)val;
  }
  __syncthreads();

  int wave = tid >> 5, lane = tid & 31;
  int tile = blockIdx.x * 8 + wave;            // 16384 tiles of 16 pixels
  int pixBase = tile * 16;
  int b  = pixBase >> 16;
  int hw = (pixBase & (kHW - 1)) + (lane & 15);
  int group = lane >> 4;

  // column base: element (b, k=0, hw) of the [B,49,H,W] planes
  const size_t colBase = (size_t)b * kN * kHW + (size_t)hw;

  // B operand: 32x16 f16 per K-step. lane = column (pixel); K = 16*group + j.
  v16h bf[2];
  #pragma unroll
  for (int s = 0; s < 2; ++s) {
    const float* cp = cl + colBase + (size_t)(16 * group + 32 * s) * kHW;
    int kbase = 32 * s + 16 * group;
    #pragma unroll
    for (int j = 0; j < 16; ++j) {
      float v = (kbase + j < kN) ? cp[(size_t)j * kHW] : 0.f;
      bf[s][j] = (_Float16)v;
    }
  }

  int mrow  = (lane & 15);
  int base8 = (lane < 16) ? 0 : 8;
  #pragma unroll
  for (int t = 0; t < 4; ++t) {
    v8f acc = {};
    #pragma unroll
    for (int s = 0; s < 2; ++s) {
      // A operand: 16x32 f16. lanes 0-15: K in 0-15; lanes 16-31: K in 16-31.
      v16h af;
      #pragma unroll
      for (int j = 0; j < 16; ++j) {
        int kk = 16 * group + j;
        af[j] = Gs[(16 * t + mrow) * 64 + 32 * s + kk];
      }
      acc = __builtin_amdgcn_wmma_f32_16x16x32_f16(
          /*neg_a=*/false, af, /*neg_b=*/false, bf[s],
          /*c_mod=*/(short)0, acc, /*reuse_a=*/false, /*reuse_b=*/false);
    }
    // D layout: lane l, reg r -> row m = 16t + r + 8*(l>>4), col = l&15.
    float* gp = ga + colBase + (size_t)(16 * t + 8 * group) * kHW;
    if (t < 3) {
      #pragma unroll
      for (int r = 0; r < 8; ++r)          // all m = 16t + r + 8g <= 47 valid
        gp[(size_t)r * kHW] = acc[r];
    } else {
      if (group == 0) gp[0] = acc[0];      // only m = 48 is a real row
    }
  }
}

// ---------------------------------------------------------------------------
// Kernel 3: per-pixel descending selection by sort key (gauss if epoch>0 else
// cl), drop the top floor(y)+1 entries, scale remainder by N/(N-y), cropped
// mean. Deterministic block partial sums.
// ---------------------------------------------------------------------------
__global__ void k_sorted_mask(const float* __restrict__ cl,
                              const float* __restrict__ ga,
                              const float* __restrict__ y,
                              const int*   __restrict__ epoch,
                              float* __restrict__ partial) {
  __shared__ float red[256];
  int idx = blockIdx.x * 256 + threadIdx.x;      // 0 .. 4*240*240-1
  int b = idx / (kInner * kInner);
  int r = idx % (kInner * kInner);
  int h = kCrop + r / kInner, w = kCrop + r % kInner;
  int hw = h * kW + w;
  bool useG = (*epoch) > 0;

  const size_t colBase = (size_t)b * kN * kHW + (size_t)hw;
  float g[kN], c[kN];
  float total = 0.f;
  #pragma unroll
  for (int k = 0; k < kN; ++k) {
    c[k] = cl[colBase + (size_t)k * kHW];
    g[k] = useG ? ga[colBase + (size_t)k * kHW] : c[k];
    total += c[k];
  }
  float yv = y[b * kHW + hw];
  int T = min(max((int)floorf(yv) + 1, 0), kN);  // masked-out count

  unsigned long long used = 0ull;
  float topsum = 0.f;
  for (int t = 0; t < T; ++t) {
    float best = -1.f, bc = 0.f;
    int bi = 0;
    #pragma unroll
    for (int k = 0; k < kN; ++k) {               // constant indices only
      bool avail = ((used >> k) & 1ull) == 0ull;
      if (avail && g[k] > best) { best = g[k]; bc = c[k]; bi = k; }
    }
    used |= 1ull << bi;
    topsum += bc;
  }
  float scale = (float)kN / ((float)kN - yv);
  float contrib = (total - topsum) * scale *
                  (1.f / ((float)kB * kN * kInner * kInner));

  red[threadIdx.x] = contrib;
  __syncthreads();
  #pragma unroll
  for (int s = 128; s > 0; s >>= 1) {
    if (threadIdx.x < s) red[threadIdx.x] += red[threadIdx.x + s];
    __syncthreads();
  }
  if (threadIdx.x == 0) partial[blockIdx.x] = red[0];
}

// ---------------------------------------------------------------------------
// Kernel 4: edge-aware smoothness, pre-scaled by ALPHA/2 and crop-region size.
// ---------------------------------------------------------------------------
__global__ void k_grad(const float* __restrict__ pred,
                       const float* __restrict__ x,
                       float* __restrict__ partial) {
  __shared__ float red[256];
  int idx = blockIdx.x * 256 + threadIdx.x;
  int b = idx / (kInner * kInner);
  int r = idx % (kInner * kInner);
  int h = kCrop + r / kInner, w = kCrop + r % kInner;
  const float* xc = x + (size_t)(b * kN + kCenter) * kHW * 3;
  const float* pr = pred + b * kHW;
  float acc = 0.f;

  if (w < kW - 1 - kCrop) {    // wq in [8, 247): 239 columns
    size_t a = ((size_t)h * kW + w) * 3;
    float ax = fabsf(xc[a + 3] - xc[a]) + fabsf(xc[a + 4] - xc[a + 1]) +
               fabsf(xc[a + 5] - xc[a + 2]);
    float wx = expf(-kEdgeC * ax * (1.f / 3.f));
    float dgx = pr[h * kW + w + 1] - pr[h * kW + w];
    acc += wx * fabsf(dgx) * (kAlpha * 0.5f / ((float)kB * kInner * 239.f));
  }
  if (h < kH - 1 - kCrop) {    // hq in [8, 247): 239 rows
    size_t a = ((size_t)h * kW + w) * 3;
    size_t d = (size_t)kW * 3;
    float ay = fabsf(xc[a + d] - xc[a]) + fabsf(xc[a + d + 1] - xc[a + 1]) +
               fabsf(xc[a + d + 2] - xc[a + 2]);
    float wy = expf(-kEdgeC * ay * (1.f / 3.f));
    float dgy = pr[(h + 1) * kW + w] - pr[h * kW + w];
    acc += wy * fabsf(dgy) * (kAlpha * 0.5f / ((float)kB * 239.f * kInner));
  }

  red[threadIdx.x] = acc;
  __syncthreads();
  #pragma unroll
  for (int s = 128; s > 0; s >>= 1) {
    if (threadIdx.x < s) red[threadIdx.x] += red[threadIdx.x + s];
    __syncthreads();
  }
  if (threadIdx.x == 0) partial[blockIdx.x] = red[0];
}

// ---------------------------------------------------------------------------
// Kernel 5: final deterministic fixed-order reduction. Writes (not adds).
// ---------------------------------------------------------------------------
__global__ void k_final(const float* __restrict__ pd,
                        const float* __restrict__ pe,
                        float* __restrict__ out) {
  __shared__ float red[256];
  float s = 0.f;
  for (int i = threadIdx.x; i < NBLK_P; i += 256) s += pd[i];
  for (int i = threadIdx.x; i < NBLK_P; i += 256) s += pe[i];
  red[threadIdx.x] = s;
  __syncthreads();
  #pragma unroll
  for (int t = 128; t > 0; t >>= 1) {
    if (threadIdx.x < t) red[threadIdx.x] += red[threadIdx.x + t];
    __syncthreads();
  }
  if (threadIdx.x == 0) out[0] = red[0];
}

extern "C" void kernel_launch(void* const* d_in, const int* in_sizes, int n_in,
                              void* d_out, int out_size, void* d_ws, size_t ws_size,
                              hipStream_t stream) {
  const float* pred  = (const float*)d_in[0];
  const float* x     = (const float*)d_in[1];
  const float* y     = (const float*)d_in[2];
  const int*   epoch = (const int*)d_in[3];

  float* ws = (float*)d_ws;
  float* cl = ws + CL_OFF;
  float* ga = ws + GA_OFF;
  float* pd = ws + PD_OFF;
  float* pe = ws + PE_OFF;

  k_color_loss<<<kB * kHW / 256, 256, 0, stream>>>(pred, x, cl);
  k_gauss_wmma<<<(kB * kHW / 16) / 8, 256, 0, stream>>>(cl, ga);
  k_sorted_mask<<<NBLK_P, 256, 0, stream>>>(cl, ga, y, epoch, pd);
  k_grad<<<NBLK_P, 256, 0, stream>>>(pred, x, pe);
  k_final<<<1, 256, 0, stream>>>(pd, pe, (float*)d_out);
}
